// DEC_71159018160361
// MI455X (gfx1250) — compile-verified, matrix operand order checked by reading, not run on hardware
//
#include <hip/hip_runtime.h>
#include <hip/hip_bf16.h>

// DEC soft-assignment for MI455X (gfx1250, wave32, WMMA).
// q = normalize_rows( 1 / (1 + ||x||^2 + ||c||^2 - 2 x.c) ), alpha=1.
// Cross term via v_wmma_f32_16x16x32_bf16; norms kept in exact fp32.

typedef __attribute__((ext_vector_type(16))) __bf16 v16bf;
typedef __attribute__((ext_vector_type(8)))  float  v8f;
typedef __attribute__((ext_vector_type(4)))  float  fvec4;
typedef __attribute__((ext_vector_type(4)))  unsigned int uvec4;

#define DEC_N   131072
#define DEC_D   512
#define DEC_K   512
#define BM      32

// ---------------------------------------------------------------------------
// Kernel 1: centers fp32 -> bf16 (into ws) + per-cluster ||c||^2 (fp32 exact)
// ---------------------------------------------------------------------------
__global__ __launch_bounds__(128)
void dec_prep_centers(const float* __restrict__ c,
                      __bf16* __restrict__ cb,
                      float* __restrict__ c2) {
    const int row = blockIdx.x;          // 512 blocks, one per cluster
    const int t   = threadIdx.x;         // 128 threads
    const float* src = c + (size_t)row * DEC_D;
    __bf16* dst      = cb + (size_t)row * DEC_D;

    float ss = 0.0f;
    #pragma unroll
    for (int i = 0; i < DEC_D / 128; ++i) {
        const int idx = t + i * 128;
        const float v = src[idx];
        ss = fmaf(v, v, ss);
        dst[idx] = (__bf16)v;
    }
    // wave32 reduction
    #pragma unroll
    for (int off = 1; off < 32; off <<= 1) ss += __shfl_xor(ss, off, 32);
    __shared__ float red[4];
    const int wid = t >> 5;
    if ((t & 31) == 0) red[wid] = ss;
    __syncthreads();
    if (t == 0) c2[row] = red[0] + red[1] + red[2] + red[3];
}

// ---------------------------------------------------------------------------
// Kernel 2: fused GEMM + epilogue.
// Block: 256 threads = 8 waves; BM=32 rows, all K=512 clusters.
// Waves: 2 row-groups x 4 col-groups; each wave -> 16 rows x 128 cols
//        = 8 wmma tiles of 16x16 (v8f accumulators).
// ---------------------------------------------------------------------------
__global__ __launch_bounds__(256)
void dec_main(const float* __restrict__ x,
              const __bf16* __restrict__ cb,
              const float* __restrict__ c2,
              float* __restrict__ out) {
    const int lane     = threadIdx.x & 31;
    const int wave     = threadIdx.x >> 5;
    const int rowGroup = wave >> 2;      // 0..1
    const int colGroup = wave & 3;       // 0..3
    const int laneHalf = lane >> 4;      // 0 or 1
    const int lane15   = lane & 15;

    const int rowBase = blockIdx.x * BM + rowGroup * 16;
    const int colBase = colGroup * 128;

    // A-fragment source: one x row per lane (lanes l and l+16 share a row,
    // holding complementary K halves per the 16-bit A VGPR layout).
    const float* xrow = x + (size_t)(rowBase + lane15) * DEC_D;

    // B-fragment sources: lane = column (cluster), contiguous K per lane.
    const __bf16* brow[8];
    #pragma unroll
    for (int t = 0; t < 8; ++t)
        brow[t] = cb + (size_t)(colBase + t * 16 + lane15) * DEC_D;

    v8f acc[8];
    #pragma unroll
    for (int t = 0; t < 8; ++t) acc[t] = (v8f){};

    float x2acc = 0.0f;   // exact fp32 ||x||^2 partial (this lane's K subset)

    for (int kk = 0; kk < DEC_D; kk += 32) {
        const int koff = kk + laneHalf * 8;

        // ---- A: 16 fp32 -> bf16 fragment (layout: elems 0..7 = K koff..+7,
        //         elems 8..15 = K koff+16..+23), accumulate ||x||^2 in fp32.
        const fvec4 a0 = *(const fvec4*)(xrow + koff);
        const fvec4 a1 = *(const fvec4*)(xrow + koff + 4);
        const fvec4 a2 = *(const fvec4*)(xrow + koff + 16);
        const fvec4 a3 = *(const fvec4*)(xrow + koff + 20);

        if (kk + 32 < DEC_D)
            __builtin_prefetch(xrow + koff + 32, 0, 3);   // global_prefetch_b8

        x2acc += a0.x*a0.x + a0.y*a0.y + a0.z*a0.z + a0.w*a0.w
               + a1.x*a1.x + a1.y*a1.y + a1.z*a1.z + a1.w*a1.w
               + a2.x*a2.x + a2.y*a2.y + a2.z*a2.z + a2.w*a2.w
               + a3.x*a3.x + a3.y*a3.y + a3.z*a3.z + a3.w*a3.w;

        v16bf av;
        av[0]  = (__bf16)a0.x; av[1]  = (__bf16)a0.y;
        av[2]  = (__bf16)a0.z; av[3]  = (__bf16)a0.w;
        av[4]  = (__bf16)a1.x; av[5]  = (__bf16)a1.y;
        av[6]  = (__bf16)a1.z; av[7]  = (__bf16)a1.w;
        av[8]  = (__bf16)a2.x; av[9]  = (__bf16)a2.y;
        av[10] = (__bf16)a2.z; av[11] = (__bf16)a2.w;
        av[12] = (__bf16)a3.x; av[13] = (__bf16)a3.y;
        av[14] = (__bf16)a3.z; av[15] = (__bf16)a3.w;

        // ---- 8 column tiles: bf16 B fragments straight from global (L2-hot)
        #pragma unroll
        for (int t = 0; t < 8; ++t) {
            union { v16bf v; uvec4 q[2]; } b;
            b.q[0] = *(const uvec4*)(brow[t] + koff);
            b.q[1] = *(const uvec4*)(brow[t] + koff + 16);
            acc[t] = __builtin_amdgcn_wmma_f32_16x16x32_bf16(
                         false, av, false, b.v,
                         (short)0, acc[t], false, false);
        }
    }

    // Complete ||x||^2: lanes l and l+16 hold complementary halves of row l%16.
    x2acc += __shfl_xor(x2acc, 16, 32);

    // ||c||^2 per column tile.
    float c2v[8];
    #pragma unroll
    for (int t = 0; t < 8; ++t) c2v[t] = c2[colBase + t * 16 + lane15];

    // Broadcast ||x||^2 to accumulator rows: acc VGPR j holds row j (lanes
    // 0-15) / row 8+j (lanes 16-31); lane r holds x2 of row r%16.
    float x2r[8];
    #pragma unroll
    for (int j = 0; j < 8; ++j)
        x2r[j] = __shfl(x2acc, j + laneHalf * 8, 32);

    // q = 1/(1 + max(dist2, 0)); accumulate row-sum partials.
    float rs[8];
    #pragma unroll
    for (int j = 0; j < 8; ++j) rs[j] = 0.0f;

    #pragma unroll
    for (int t = 0; t < 8; ++t) {
        #pragma unroll
        for (int j = 0; j < 8; ++j) {
            float d = x2r[j] + c2v[t] - 2.0f * acc[t][j];
            d = fmaxf(d, 0.0f);
            const float q = 1.0f / (1.0f + d);
            acc[t][j] = q;
            rs[j] += q;
        }
    }

    // Reduce row sums over the 16 lanes of each half (columns within wave).
    #pragma unroll
    for (int j = 0; j < 8; ++j) {
        rs[j] += __shfl_xor(rs[j], 1, 32);
        rs[j] += __shfl_xor(rs[j], 2, 32);
        rs[j] += __shfl_xor(rs[j], 4, 32);
        rs[j] += __shfl_xor(rs[j], 8, 32);
    }

    // Cross-wave (col-group) reduction through LDS.
    __shared__ float s_sums[4][32];
    #pragma unroll
    for (int j = 0; j < 8; ++j) {
        if (lane15 == j)
            s_sums[colGroup][rowGroup * 16 + j + laneHalf * 8] = rs[j];
    }
    __syncthreads();

    // Normalize and write q (coalesced 16-lane groups per tile).
    #pragma unroll
    for (int j = 0; j < 8; ++j) {
        const int rIdx = rowGroup * 16 + j + laneHalf * 8;
        const float tot = s_sums[0][rIdx] + s_sums[1][rIdx]
                        + s_sums[2][rIdx] + s_sums[3][rIdx];
        const float inv = 1.0f / tot;
        const size_t row = (size_t)(blockIdx.x * BM) + (size_t)rIdx;
        float* orow = out + row * DEC_K + colBase + lane15;
        #pragma unroll
        for (int t = 0; t < 8; ++t)
            orow[t * 16] = acc[t][j] * inv;
    }
}

// ---------------------------------------------------------------------------
extern "C" void kernel_launch(void* const* d_in, const int* in_sizes, int n_in,
                              void* d_out, int out_size, void* d_ws, size_t ws_size,
                              hipStream_t stream) {
    const float* x  = (const float*)d_in[0];                 // (131072, 512)
    const float* cc = (const float*)d_in[1];                 // (512, 512)
    float* out      = (float*)d_out;                         // (131072, 512)

    __bf16* cb = (__bf16*)d_ws;                              // 512*512*2 = 512 KB
    float*  c2 = (float*)((char*)d_ws + (size_t)DEC_K * DEC_D * sizeof(__bf16));

    dec_prep_centers<<<DEC_K, 128, 0, stream>>>(cc, cb, c2);
    dec_main<<<DEC_N / BM, 256, 0, stream>>>(x, cb, c2, out);
}
